// SlidingWindowAttention_62947040690413
// MI455X (gfx1250) — matrix operand order repaired and checked by reading, not compiled, not run on hardware
//
#include <hip/hip_runtime.h>

typedef __attribute__((ext_vector_type(16))) __bf16 v16bf;
typedef __attribute__((ext_vector_type(8)))  float  v8f;
typedef __attribute__((ext_vector_type(4)))  int    v4i;

// ---------------------------------------------------------------------------
// CDNA5 async global->LDS copy (ASYNCcnt-tracked), with sync fallback.
// Builtin signature (from hipcc diagnostic): param0 = addrspace(1) int4*,
// so pass exactly-typed AS1/AS3 int4 pointers.
// ---------------------------------------------------------------------------
#if defined(__has_builtin)
#if __has_builtin(__builtin_amdgcn_global_load_async_to_lds_b128)
#define USE_ASYNC_LDS 1
#endif
#endif

typedef __attribute__((address_space(1))) v4i GInt4;
typedef __attribute__((address_space(3))) v4i LInt4;

__device__ inline void cp_g2l_b128(void* lds, const void* g) {
#ifdef USE_ASYNC_LDS
  __builtin_amdgcn_global_load_async_to_lds_b128(
      (GInt4*)g, (LInt4*)lds, 0, 0);
#else
  *(uint4*)lds = *(const uint4*)g;
#endif
}

__device__ inline void wait_async() {
#ifdef USE_ASYNC_LDS
#if __has_builtin(__builtin_amdgcn_s_wait_asynccnt)
  __builtin_amdgcn_s_wait_asynccnt(0);
#endif
#endif
}

// ---------------------------------------------------------------------------
// Fragment loaders (wave32 WMMA 16x16x32 bf16 layouts per CDNA5 ISA §7.12.2)
// A 16x32 (MxK): lane = M row, half = lane/16:
//   element i (vg=i/2): k = (vg/4)*16 + half*8 + (vg%4)*2 (+pair)
// B 32x16 (KxN): lane = N col, k = (vg/4)*16 + half*8 + (vg%4)*2 (+pair)
// C/D 16x16: lane -> n = lane%16 ; vgpr r -> m = r + 8*(lane/16)
// Both loaders read 2x16 contiguous bytes -> compiler emits ds_load_b128 x2.
// ---------------------------------------------------------------------------
__device__ inline v16bf load_a_frag(const __bf16* row, int half) {
  union { v16bf v; unsigned u[8]; } r;
#pragma unroll
  for (int vg = 0; vg < 8; ++vg) {
    int k = ((vg >> 2) << 4) + half * 8 + ((vg & 3) << 1);
    r.u[vg] = *(const unsigned*)(row + k);
  }
  return r.v;
}

__device__ inline v16bf load_b_frag(const __bf16* col, int half) {
  union { v16bf v; unsigned u[8]; } r;
#pragma unroll
  for (int vg = 0; vg < 8; ++vg) {
    int k = ((vg >> 2) << 4) + half * 8 + ((vg & 3) << 1);
    r.u[vg] = *(const unsigned*)(col + k);
  }
  return r.v;
}

// ---------------------------------------------------------------------------
// fp32 -> bf16 cast with optional column packing (used to fuse Wq|Wk|Wv)
// ---------------------------------------------------------------------------
__global__ void cast_pack(const float* __restrict__ in, __bf16* __restrict__ out,
                          int n, int cols, int ostride, int ocol0) {
  for (int idx = blockIdx.x * blockDim.x + threadIdx.x; idx < n;
       idx += gridDim.x * blockDim.x) {
    int r = idx / cols, c = idx % cols;
    out[(size_t)r * ostride + ocol0 + c] = (__bf16)in[idx];
  }
}

// ---------------------------------------------------------------------------
// GEMM core: C(128x64 tile) = A[M,KD] @ B[KD,ND], bf16 in, f32 acc.
// 256 threads = 8 waves; wave w owns rows [w*16, w*16+16), all 64 cols.
// Double-buffered LDS: next K-step's A tile streams in via async-to-LDS
// while WMMAs consume the current one.
// ---------------------------------------------------------------------------
template <int KD, int ND>
__device__ inline void gemm_tile(const __bf16* __restrict__ A,
                                 const __bf16* __restrict__ Bw,
                                 int mBase, int nBase,
                                 __bf16 (&As)[2][128][40],
                                 __bf16 (&Bs)[2][64][40], v8f (&acc)[4]) {
  int tid = threadIdx.x, lane = tid & 31, wave = tid >> 5;
  int half = lane >> 4, ml = lane & 15;

  auto stage = [&](int bufi, int k0) {
    // A tile 128x32, verbatim copy -> async global->LDS, 16B granules
#pragma unroll
    for (int p = 0; p < 2; ++p) {
      int e = p * 2048 + tid * 8;
      int r = e >> 5, c = e & 31;
      cp_g2l_b128(&As[bufi][r][c], A + (size_t)(mBase + r) * KD + k0 + c);
    }
    // B tile 32x64 transposed -> Bs[n][k] (k-pairs contiguous for frags)
    int kk = tid >> 3, n = (tid & 7) * 8;
    uint4 d = *(const uint4*)(Bw + (size_t)(k0 + kk) * ND + nBase + n);
    const __bf16* dv = (const __bf16*)&d;
#pragma unroll
    for (int j = 0; j < 8; ++j) Bs[bufi][n + j][kk] = dv[j];
    if (k0 + 32 < KD)
      __builtin_prefetch(Bw + (size_t)(k0 + 32 + kk) * ND + nBase + n, 0, 0);
  };

  stage(0, 0);
  wait_async();
  __syncthreads();

  int buf = 0;
  for (int k0 = 0; k0 < KD; k0 += 32) {
    if (k0 + 32 < KD) stage(buf ^ 1, k0 + 32);  // overlap with WMMA below
    v16bf a = load_a_frag(&As[buf][wave * 16 + ml][0], half);
#pragma unroll
    for (int nt = 0; nt < 4; ++nt) {
      v16bf b = load_b_frag(&Bs[buf][nt * 16 + ml][0], half);
      acc[nt] = __builtin_amdgcn_wmma_f32_16x16x32_bf16(
          false, a, false, b, (short)0, acc[nt], false, false);
    }
    wait_async();
    __syncthreads();
    buf ^= 1;
  }
}

// ---------------------------------------------------------------------------
// Fused QKV projection: xh[4096,2048] @ Wqkv[2048,3072] (+bias), scatter:
//   cols [0,2048)    -> Qb row-major [4096,2048] bf16
//   cols [2048,2560) -> Kb [b,g,s,128] bf16
//   cols [2560,3072) -> Vb [b,g,s,128] bf16
// ---------------------------------------------------------------------------
__global__ __launch_bounds__(256) void gemm_qkv(
    const __bf16* __restrict__ xh, const __bf16* __restrict__ Wqkv,
    const float* __restrict__ bq, const float* __restrict__ bk,
    const float* __restrict__ bv, __bf16* __restrict__ Qb,
    __bf16* __restrict__ Kb, __bf16* __restrict__ Vb) {
  __shared__ __bf16 As[2][128][40];
  __shared__ __bf16 Bs[2][64][40];
  v8f acc[4] = {};
  int mBase = blockIdx.x * 128, nBase = blockIdx.y * 64;
  gemm_tile<2048, 3072>(xh, Wqkv, mBase, nBase, As, Bs, acc);

  int tid = threadIdx.x, lane = tid & 31, wave = tid >> 5;
  int half = lane >> 4, nl = lane & 15;
#pragma unroll
  for (int nt = 0; nt < 4; ++nt) {
    int n = nBase + nt * 16 + nl;
#pragma unroll
    for (int r = 0; r < 8; ++r) {
      int m = mBase + wave * 16 + r + 8 * half;  // m = b*2048 + s
      float v = acc[nt][r];
      if (n < 2048) {
        Qb[(size_t)m * 2048 + n] = (__bf16)(v + bq[n]);
      } else if (n < 2560) {
        int c = n - 2048, g = c >> 7, hd = c & 127;
        int b = m >> 11, s = m & 2047;
        Kb[(((size_t)(b * 4 + g)) * 2048 + s) * 128 + hd] = (__bf16)(v + bk[c]);
      } else {
        int c = n - 2560, g = c >> 7, hd = c & 127;
        int b = m >> 11, s = m & 2047;
        Vb[(((size_t)(b * 4 + g)) * 2048 + s) * 128 + hd] = (__bf16)(v + bv[c]);
      }
    }
  }
}

// ---------------------------------------------------------------------------
// Sliding-window GQA attention, flash-style online softmax.
// One wave (32 threads) per (batch, head, 16-query tile).
// 9 chunks of 32 keys cover [i0-256, i0+31]; masking is data-only so EXEC
// stays all-1s for WMMA. 16 WMMAs per chunk: 8 QK^T + 8 PV.
// K chunk staged via async global->LDS (verbatim); V staged transposed.
// ---------------------------------------------------------------------------
__global__ __launch_bounds__(32) void attn(const __bf16* __restrict__ Qb,
                                           const __bf16* __restrict__ Kb,
                                           const __bf16* __restrict__ Vb,
                                           __bf16* __restrict__ Ob) {
  __shared__ __bf16 Ks[32][136];  // K chunk:  [key][hd], 272B rows (16B mult)
  __shared__ __bf16 Vt[128][40];  // V^T chunk: [hd][key]
  __shared__ __bf16 Pl[16][40];   // probabilities: [query][key]

  int bid = blockIdx.x;
  int qt = bid & 127;
  int h = (bid >> 7) & 15;
  int b = bid >> 11;
  int g = h >> 2;  // QPER = 4
  int i0 = qt << 4;
  int lane = threadIdx.x & 31;
  int half = lane >> 4, nl = lane & 15;

  // Q A-fragments (16 queries x 128 hd = 4 K-chunks of 32), loaded once.
  const __bf16* qrow = Qb + ((size_t)(b * 2048 + i0 + nl)) * 2048 + h * 128;
  v16bf qa[4];
#pragma unroll
  for (int kk = 0; kk < 4; ++kk) qa[kk] = load_a_frag(qrow + kk * 32, half);

  const __bf16* Kh = Kb + ((size_t)(b * 4 + g)) * 2048 * 128;
  const __bf16* Vh = Vb + ((size_t)(b * 4 + g)) * 2048 * 128;

  v8f o[8] = {};
  float rmax[8], rsum[8];
#pragma unroll
  for (int r = 0; r < 8; ++r) { rmax[r] = -1e30f; rsum[r] = 0.f; }

  const float scale = 0.08838834764831845f;  // 1/sqrt(128)

  for (int c = 0; c < 9; ++c) {
    int j0 = i0 - 256 + c * 32;
    __syncthreads();  // prior-iteration LDS reads done before overwrite
    int j = j0 + lane;
    int jm = j < 0 ? 0 : (j > 2047 ? 2047 : j);
    // Stage K chunk (lane <-> key): async verbatim copy, 16 x b128
    {
      const __bf16* krow = Kh + (size_t)jm * 128;
#pragma unroll
      for (int q = 0; q < 16; ++q)
        cp_g2l_b128(&Ks[lane][q * 8], krow + q * 8);
    }
    // Stage V^T (lane <-> key): transpose scatter
    {
      const __bf16* vrow = Vh + (size_t)jm * 128;
#pragma unroll
      for (int q = 0; q < 16; ++q) {
        uint4 d = *(const uint4*)(vrow + q * 8);
        const __bf16* dv = (const __bf16*)&d;
#pragma unroll
        for (int e = 0; e < 8; ++e) Vt[q * 8 + e][lane] = dv[e];
      }
    }
    wait_async();
    __syncthreads();  // Ks/Vt visible across lanes

    // Scores: S = Q(16x128) @ K_chunk^T(128x32) as two 16-key sub-tiles
    v8f sc0 = {}, sc1 = {};
#pragma unroll
    for (int kk = 0; kk < 4; ++kk) {
      v16bf kf0 = load_b_frag(&Ks[nl][kk * 32], half);
      sc0 = __builtin_amdgcn_wmma_f32_16x16x32_bf16(
          false, qa[kk], false, kf0, (short)0, sc0, false, false);
      v16bf kf1 = load_b_frag(&Ks[16 + nl][kk * 32], half);
      sc1 = __builtin_amdgcn_wmma_f32_16x16x32_bf16(
          false, qa[kk], false, kf1, (short)0, sc1, false, false);
    }
    // Online softmax per row (row m = r + 8*half lives in this half's lanes)
#pragma unroll
    for (int r = 0; r < 8; ++r) {
      int i = i0 + r + 8 * half;
      int ja = j0 + nl, jb = j0 + 16 + nl;
      bool k0v = (ja >= 0) && (ja <= i) && (ja > i - 256);
      bool k1v = (jb >= 0) && (jb <= i) && (jb > i - 256);
      float x0 = k0v ? sc0[r] * scale : -1e30f;
      float x1 = k1v ? sc1[r] * scale : -1e30f;
      float mx = fmaxf(x0, x1);
#pragma unroll
      for (int off = 1; off < 16; off <<= 1)
        mx = fmaxf(mx, __shfl_xor(mx, off, 32));
      float nm = fmaxf(rmax[r], mx);
      float al = __expf(rmax[r] - nm);
      float p0 = k0v ? __expf(x0 - nm) : 0.f;
      float p1 = k1v ? __expf(x1 - nm) : 0.f;
      float sm = p0 + p1;
#pragma unroll
      for (int off = 1; off < 16; off <<= 1) sm += __shfl_xor(sm, off, 32);
      rsum[r] = rsum[r] * al + sm;
      rmax[r] = nm;
#pragma unroll
      for (int nt = 0; nt < 8; ++nt) o[nt][r] *= al;
      Pl[r + 8 * half][nl] = (__bf16)p0;
      Pl[r + 8 * half][16 + nl] = (__bf16)p1;
    }
    __syncthreads();
    // O += P(16x32) @ V(32x128): one A-frag, 8 hd tiles
    v16bf pa = load_a_frag(&Pl[nl][0], half);
#pragma unroll
    for (int nt = 0; nt < 8; ++nt) {
      v16bf vf = load_b_frag(&Vt[nt * 16 + nl][0], half);
      o[nt] = __builtin_amdgcn_wmma_f32_16x16x32_bf16(
          false, pa, false, vf, (short)0, o[nt], false, false);
    }
  }
  // Normalize and store bf16 O in [b, s, h*128+hd] layout for the out-GEMM
#pragma unroll
  for (int r = 0; r < 8; ++r) {
    float inv = 1.f / rsum[r];
    int m = i0 + r + 8 * half;
    __bf16* orow = Ob + ((size_t)(b * 2048 + m)) * 2048 + h * 128;
#pragma unroll
    for (int nt = 0; nt < 8; ++nt)
      orow[nt * 16 + nl] = (__bf16)(o[nt][r] * inv);
  }
}

// ---------------------------------------------------------------------------
// Output projection: Ob[4096,2048] @ Wo[2048,2048] + bo -> fp32 out
// ---------------------------------------------------------------------------
__global__ __launch_bounds__(256) void gemm_out(const __bf16* __restrict__ Ob,
                                                const __bf16* __restrict__ Woh,
                                                const float* __restrict__ bo,
                                                float* __restrict__ out) {
  __shared__ __bf16 As[2][128][40];
  __shared__ __bf16 Bs[2][64][40];
  v8f acc[4] = {};
  int mBase = blockIdx.x * 128, nBase = blockIdx.y * 64;
  gemm_tile<2048, 2048>(Ob, Woh, mBase, nBase, As, Bs, acc);

  int tid = threadIdx.x, lane = tid & 31, wave = tid >> 5;
  int half = lane >> 4, nl = lane & 15;
#pragma unroll
  for (int nt = 0; nt < 4; ++nt) {
    int n = nBase + nt * 16 + nl;
#pragma unroll
    for (int r = 0; r < 8; ++r) {
      int m = mBase + wave * 16 + r + 8 * half;
      out[(size_t)m * 2048 + n] = acc[nt][r] + bo[n];
    }
  }
}

// ---------------------------------------------------------------------------
// Host launcher. Workspace layout (bytes):
//   [0,        16.8M) xh  (bf16 x)        -- reused as Ob after gemm_qkv
//   [16.8M,    29.4M) Wqkv bf16 [2048,3072]
//   [29.4M,    37.7M) Woh  bf16 [2048,2048]
//   [37.7M,    54.5M) Qb   bf16 [4096,2048]
//   [54.5M,    58.7M) Kb   bf16 [2,4,2048,128]
//   [58.7M,    62.9M) Vb   bf16 [2,4,2048,128]
// ---------------------------------------------------------------------------
extern "C" void kernel_launch(void* const* d_in, const int* in_sizes, int n_in,
                              void* d_out, int out_size, void* d_ws,
                              size_t ws_size, hipStream_t stream) {
  const float* x  = (const float*)d_in[0];
  const float* Wq = (const float*)d_in[1];
  const float* bq = (const float*)d_in[2];
  const float* Wk = (const float*)d_in[3];
  const float* bk = (const float*)d_in[4];
  const float* Wv = (const float*)d_in[5];
  const float* bv = (const float*)d_in[6];
  const float* Wo = (const float*)d_in[7];
  const float* bo = (const float*)d_in[8];
  float* out = (float*)d_out;

  char* ws = (char*)d_ws;
  __bf16* xh   = (__bf16*)(ws + 0);
  __bf16* Wqkv = (__bf16*)(ws + 16777216);
  __bf16* Woh  = (__bf16*)(ws + 29360128);
  __bf16* Qb   = (__bf16*)(ws + 37748736);
  __bf16* Kb   = (__bf16*)(ws + 54525952);
  __bf16* Vb   = (__bf16*)(ws + 58720256);
  __bf16* Ob   = (__bf16*)(ws + 0);  // alias xh: dead after gemm_qkv

  cast_pack<<<2048, 256, 0, stream>>>(x,  xh,   8388608, 2048, 2048, 0);
  cast_pack<<<1024, 256, 0, stream>>>(Wq, Wqkv, 4194304, 2048, 3072, 0);
  cast_pack<<<256,  256, 0, stream>>>(Wk, Wqkv, 1048576, 512,  3072, 2048);
  cast_pack<<<256,  256, 0, stream>>>(Wv, Wqkv, 1048576, 512,  3072, 2560);
  cast_pack<<<1024, 256, 0, stream>>>(Wo, Woh,  4194304, 2048, 2048, 0);

  gemm_qkv<<<dim3(32, 48), 256, 0, stream>>>(xh, Wqkv, bq, bk, bv, Qb, Kb, Vb);
  attn<<<4096, 32, 0, stream>>>(Qb, Kb, Vb, Ob);
  gemm_out<<<dim3(32, 32), 256, 0, stream>>>(Ob, Woh, bo, out);
}
